// myQBCResNet_36498632081878
// MI455X (gfx1250) — matrix operand order repaired and checked by reading, not compiled, not run on hardware
//
#include <hip/hip_runtime.h>

// ---------------------------------------------------------------------------
// CDNA5 (gfx1250) fused BC-ResNet forward.
//  * int16 fixed-point activation storage (AQ grid, exact) -> halves HBM traffic
//  * pointwise 16x16 conv via v_wmma_f32_16x16x32_f16 (K padded 16->32),
//    4 WMMA tiles per wave sharing one A fragment
//  * ymean slab staged to LDS via TDM tensor_load_to_lds (TENSORcnt) when the
//    builtin exists; cooperative-copy fallback otherwise
// ---------------------------------------------------------------------------

typedef __attribute__((ext_vector_type(16))) _Float16 v16h;
typedef __attribute__((ext_vector_type(8)))  float    v8f;
typedef unsigned int u32x4 __attribute__((ext_vector_type(4)));
typedef int          i32x8 __attribute__((ext_vector_type(8)));
typedef int          i32x4 __attribute__((ext_vector_type(4)));

#if defined(__HIP_DEVICE_COMPILE__) && defined(__has_builtin)
#if __has_builtin(__builtin_amdgcn_tensor_load_to_lds) && \
    __has_builtin(__builtin_amdgcn_s_wait_tensorcnt)
#define HAVE_TDM 1
#endif
#endif
#ifndef HAVE_TDM
#define HAVE_TDM 0
#endif

__device__ __forceinline__ float clampf_(float x, float lo, float hi) {
  return fminf(fmaxf(x, lo), hi);
}
// fixed-point quantize: round-to-nearest-even on 2^-frac grid, saturate.
__device__ __forceinline__ float fixq_(float x, float s, float invs,
                                       float qmin, float qmax) {
  return clampf_(rintf(x * s) * invs, qmin, qmax);
}
__device__ __forceinline__ float fixq_AQ(float x) {   // (1,16,9)
  return fixq_(x, 512.f, 1.f / 512.f, -64.f, 32767.f / 512.f);
}
__device__ __forceinline__ float fixq_AQS(float x) {  // (1,8,4)
  return fixq_(x, 16.f, 1.f / 16.f, -8.f, 127.f / 16.f);
}
__device__ __forceinline__ float fixq_W(float x) {    // (1,8,6)
  return fixq_(x, 64.f, 1.f / 64.f, -2.f, 127.f / 64.f);
}
__device__ __forceinline__ float fixq_BN(float x) { return fixq_AQ(x); }

// AQ-grid value <-> int16 (exact: AQ is 16-bit fixed point, frac 9)
__device__ __forceinline__ short packAQ(float x) {
  return (short)clampf_(rintf(x * 512.f), -32768.f, 32767.f);
}
__device__ __forceinline__ float unpackAQ(short s) {
  return (float)s * (1.f / 512.f);
}

// ---------------------------------------------------------------------------
// conv1: (256,1,40,500) -> (256,16,20,500), 5x5, stride (2,1), pad 2, + bias,
// input prescaled by 1/64, output fixq(AQ) packed to int16.
// ---------------------------------------------------------------------------
__global__ void k_conv1(const float* __restrict__ x, const float* __restrict__ w,
                        const float* __restrict__ b, short* __restrict__ out) {
  long gid = (long)blockIdx.x * blockDim.x + threadIdx.x;
  int t = (int)(gid % 512);
  long r = gid / 512;
  int f = (int)(r % 20); r /= 20;
  int co = (int)(r % 16); r /= 16;
  int n = (int)r;
  if (t >= 500 || n >= 256) return;
  const float* xp = x + (long)n * 40 * 500;
  __builtin_prefetch(xp + (2 * f) * 500 + t + 256, 0, 1);  // global_prefetch_b8
  float acc = fixq_W(b[co]);
#pragma unroll
  for (int kf = 0; kf < 5; ++kf) {
    int fi = 2 * f + kf - 2;
    if (fi < 0 || fi >= 40) continue;
#pragma unroll
    for (int kt = 0; kt < 5; ++kt) {
      int ti = t + kt - 2;
      if (ti < 0 || ti >= 500) continue;
      acc += (xp[fi * 500 + ti] * (1.f / 64.f)) * fixq_W(w[(co * 5 + kf) * 5 + kt]);
    }
  }
  out[((long)(n * 16 + co) * 20 + f) * 500 + t] = packAQ(fixq_AQ(acc));
}

// ---------------------------------------------------------------------------
// Per block, stage 1: depthwise freq conv (3x1, stride s, pad 1) + fixq(AQ)
// + sub-spectral norm + fixq(AQ); writes aux (AQS-requantized for T blocks)
// and the freq-mean (fixq AQ).  grid: (t_tile=4, c=16, n=256), block 128.
// ---------------------------------------------------------------------------
__global__ void k_freqdw(const short* __restrict__ in, short* __restrict__ aux,
                         float* __restrict__ ymean, const float* __restrict__ fdw,
                         const float* __restrict__ sg, const float* __restrict__ sb,
                         const float* __restrict__ sm, const float* __restrict__ sv,
                         int FrIn, int FrOut, int stride, int kindT) {
  int t = blockIdx.x * 128 + threadIdx.x;
  if (t >= 500) return;
  int c = blockIdx.y, n = blockIdx.z;
  const short* ip = in + (long)(n * 16 + c) * FrIn * 500;
  short* op = aux + (long)(n * 16 + c) * FrOut * 500;
  float w0 = fixq_W(fdw[c * 3 + 0]);
  float w1 = fixq_W(fdw[c * 3 + 1]);
  float w2 = fixq_W(fdw[c * 3 + 2]);
  int sub = FrOut / 5;  // freq rows per sub-band
  float accm = 0.f;
  for (int f = 0; f < FrOut; ++f) {
    int fc = stride * f;
    float a = unpackAQ(ip[fc * 500 + t]) * w1;
    if (fc - 1 >= 0)   a += unpackAQ(ip[(fc - 1) * 500 + t]) * w0;
    if (fc + 1 < FrIn) a += unpackAQ(ip[(fc + 1) * 500 + t]) * w2;
    a = fixq_AQ(a);
    int sc = c * 5 + f / sub;  // folded SSN channel
    float mult = sg[sc] / sqrtf(sv[sc] + 1e-5f);
    float o = fixq_AQ(a * fixq_BN(mult) + fixq_BN(sb[sc] - sm[sc] * mult));
    accm += o;
    op[f * 500 + t] = packAQ(kindT ? fixq_AQS(o) : o);
  }
  ymean[(long)(n * 16 + c) * 500 + t] = fixq_AQ(accm / (float)FrOut);
}

// ---------------------------------------------------------------------------
// Per block, stage 2: one workgroup per n (8 waves). TDM stages the 16x500
// ymean slab into LDS (tensor_load_to_lds, TENSORcnt), then each wave runs
// dilated depthwise temporal conv + BN + ReLU (all fixq'd) and 4 pointwise
// 16x16 GEMM tiles via v_wmma_f32_16x16x32_f16 sharing one A fragment.
// ---------------------------------------------------------------------------
__global__ void __launch_bounds__(256)
k_temporal_pw(const float* __restrict__ ymean, float* __restrict__ pwout,
              const float* __restrict__ tdw, const float* __restrict__ pw,
              const float* __restrict__ bg, const float* __restrict__ bb,
              const float* __restrict__ bm, const float* __restrict__ bv,
              int dil) {
  __shared__ float lys[16 * 500];           // 31.25 KB of 320 KB/WGP
  __shared__ float s_scale[16], s_bias[16], s_wq[48];
  int tid = threadIdx.x;
  int n = blockIdx.x;
  const float* yb = ymean + (long)n * 16 * 500;

  if (tid < 16) {
    float mult = bg[tid] / sqrtf(bv[tid] + 1e-5f);
    s_scale[tid] = fixq_BN(mult);
    s_bias[tid]  = fixq_BN(bb[tid] - bm[tid] * mult);
  }
  if (tid < 48) s_wq[tid] = fixq_W(tdw[tid]);

#if HAVE_TDM
  if (tid < 32) {  // wave 0 issues one TDM 2D tile copy: 500x16 f32 -> LDS
    unsigned long long ga = (unsigned long long)(uintptr_t)yb;
    unsigned lé = (unsigned)(uintptr_t)lys;  // low 32 bits = LDS byte offset
    u32x4 g0;
    g0[0] = 1u;                               // count=1, user descriptor
    g0[1] = lé;                               // lds_addr
    g0[2] = (unsigned)(ga & 0xffffffffu);     // global_addr[31:0]
    g0[3] = (unsigned)((ga >> 32) & 0x1ffffffu) | (2u << 30);  // addr[56:32]|type=2
    i32x8 g1;
    g1[0] = (2 << 16);          // workgroup_mask=0, data_size=4B
    g1[1] = (500 << 16);        // tensor_dim0[15:0]=500 (atomic_barrier_addr=0)
    g1[2] = (16 << 16);         // tensor_dim0[31:16]=0, tensor_dim1[15:0]=16
    g1[3] = (500 << 16);        // tensor_dim1[31:16]=0, tile_dim0=500
    g1[4] = 16;                 // tile_dim1=16, tile_dim2=0
    g1[5] = 500;                // tensor_dim0_stride[31:0]=500
    g1[6] = 0;                  // stride0[47:32]=0, tensor_dim1_stride[15:0]=0
    g1[7] = 0;
    i32x4 g2; g2[0] = 0; g2[1] = 0; g2[2] = 0; g2[3] = 0;  // 2D: groups 2/3 unused
    i32x4 g3; g3[0] = 0; g3[1] = 0; g3[2] = 0; g3[3] = 0;
#if defined(__clang_major__) && (__clang_major__ >= 23)
    i32x8 g4; g4[0]=0; g4[1]=0; g4[2]=0; g4[3]=0; g4[4]=0; g4[5]=0; g4[6]=0; g4[7]=0;
    __builtin_amdgcn_tensor_load_to_lds(g0, g1, g2, g3, g4, 0);
#else
    __builtin_amdgcn_tensor_load_to_lds(g0, g1, g2, g3, 0);
#endif
    __builtin_amdgcn_s_wait_tensorcnt(0);
  }
#else
  for (int i = tid; i < 16 * 500; i += 256) lys[i] = yb[i];
#endif
  __syncthreads();

  int wave = tid >> 5, lane = tid & 31;
  // A fragment (16x32 f16), built once per wave: row M = lane&15 (=co);
  // contiguous layout K = Kbase + h, Kbase = lane<16?0:16; K>=16 pad.
  union { v16h v; _Float16 h[16]; } afrag;
  int co = lane & 15;
  int kbase = (lane & 16) ? 16 : 0;
#pragma unroll
  for (int h = 0; h < 16; ++h) {
    int K = kbase + h;
    afrag.h[h] = (_Float16)((K < 16) ? fixq_W(pw[co * 16 + K]) : 0.f);
  }
  bool low = lane < 16;
  float* ob = pwout + (long)n * 16 * 500;
  int obase = low ? 0 : 8;

#pragma unroll
  for (int st = 0; st < 4; ++st) {          // 8 waves x 4 tiles = 32 tiles >= 500/16
    int t0 = (wave * 4 + st) * 16;
    int tc = t0 + (lane & 15);
    // B fragment (32x16 f16): lane N = lane&15; lanes 0-15 hold K=0..15 (=ci),
    // lanes 16-31 hold the zero-pad rows K=16..31.
    union { v16h v; _Float16 h[16]; } bfrag;
#pragma unroll
    for (int ci = 0; ci < 16; ++ci) {
      float z = 0.f;
      if (low && tc < 500) {
        const float* yr = lys + ci * 500;
        float acc = yr[tc] * s_wq[ci * 3 + 1];
        int tm = tc - dil, tp = tc + dil;
        if (tm >= 0)  acc += yr[tm] * s_wq[ci * 3 + 0];
        if (tp < 500) acc += yr[tp] * s_wq[ci * 3 + 2];
        acc = fixq_AQ(acc);
        acc = fixq_AQ(acc * s_scale[ci] + s_bias[ci]);
        z = fixq_AQ(fmaxf(acc, 0.f));       // relu then requant
      }
      bfrag.h[ci] = (_Float16)z;
    }
    union { v8f v; float f[8]; } cacc;
#pragma unroll
    for (int r = 0; r < 8; ++r) cacc.f[r] = 0.f;
    cacc.v = __builtin_amdgcn_wmma_f32_16x16x32_f16(
        /*neg_a=*/false, afrag.v, /*neg_b=*/false, bfrag.v,
        /*c_mod=*/(short)0, cacc.v, /*reuse_a=*/false, /*reuse_b=*/false);
    // D layout: VGPR r -> (co = r + (lane<16?0:8), t = t0 + (lane&15))
    int ot = t0 + (lane & 15);
    if (ot < 500) {
#pragma unroll
      for (int r = 0; r < 8; ++r)
        ob[(obase + r) * 500 + ot] = fixq_AQ(cacc.f[r]);
    }
  }
}

// ---------------------------------------------------------------------------
// Per block, stage 3: residual combine + ReLU, writes next block's input.
//  B: relu(fixq(fixq(pw + aux, AQS) + x, AQ));  T: relu(fixq(aux + pw, AQ))
// ---------------------------------------------------------------------------
__global__ void k_residual(const short* __restrict__ aux,
                           const float* __restrict__ pwout,
                           short* __restrict__ io, int FrIn, int FrOut, int kindT) {
  long gid = (long)blockIdx.x * blockDim.x + threadIdx.x;
  long total = (long)256 * 16 * FrOut * 500;
  if (gid >= total) return;
  int t = (int)(gid % 500);
  long r = gid / 500;
  int f = (int)(r % FrOut); r /= FrOut;
  int c = (int)(r % 16); r /= 16;
  int n = (int)r;
  float pv = pwout[(long)(n * 16 + c) * 500 + t];
  float av = unpackAQ(aux[((long)(n * 16 + c) * FrOut + f) * 500 + t]);
  float o;
  if (kindT) {
    o = fixq_AQ(av + pv);
  } else {  // B blocks always have FrIn == FrOut (stride 1)
    float xv = unpackAQ(io[((long)(n * 16 + c) * FrIn + f) * 500 + t]);
    o = fixq_AQ(fixq_AQS(pv + av) + xv);
  }
  io[((long)(n * 16 + c) * FrOut + f) * 500 + t] = packAQ(fmaxf(o, 0.f));
}

// ---------------------------------------------------------------------------
// Final: mean over T of (256,16,5,500), then 12x80 FC (conv4). One WG per n.
// ---------------------------------------------------------------------------
__global__ void k_final(const short* __restrict__ act,
                        const float* __restrict__ w4, float* __restrict__ out) {
  __shared__ float mean[80];
  int n = blockIdx.x, tid = threadIdx.x;
  if (tid < 80) {
    const short* p = act + ((long)n * 80 + tid) * 500;  // rows = c*5+f
    float s = 0.f;
    for (int t = 0; t < 500; ++t) s += unpackAQ(p[t]);
    mean[tid] = s * (1.f / 500.f);
  }
  __syncthreads();
  if (tid < 12) {
    float acc = 0.f;
#pragma unroll
    for (int j = 0; j < 80; ++j) acc += mean[j] * fixq_W(w4[tid * 80 + j]);
    out[n * 12 + tid] = acc;
  }
}

// ---------------------------------------------------------------------------
// Host driver. d_in flat order: x, conv1_w, conv1_b,
//   12 x {fdw, tdw, pw, ssn_g, ssn_b, ssn_m, ssn_v, bn_g, bn_b, bn_m, bn_v},
//   conv4_w.
// ---------------------------------------------------------------------------
extern "C" void kernel_launch(void* const* d_in, const int* in_sizes, int n_in,
                              void* d_out, int out_size, void* d_ws, size_t ws_size,
                              hipStream_t stream) {
  (void)in_sizes; (void)n_in; (void)out_size; (void)ws_size;
  const float* x       = (const float*)d_in[0];
  const float* conv1_w = (const float*)d_in[1];
  const float* conv1_b = (const float*)d_in[2];
  const float* conv4_w = (const float*)d_in[3 + 12 * 11];

  const long ACT_ELEMS = 256L * 16 * 20 * 500;   // 40,960,000
  const long TVEC_ELEMS = 256L * 16 * 500;       //  2,048,000
  char* ws = (char*)d_ws;
  short* bufA  = (short*)ws;                                   // ping (int16 AQ)
  short* bufB  = (short*)(ws + ACT_ELEMS * sizeof(short));     // aux  (int16)
  float* ymean = (float*)(ws + 2 * ACT_ELEMS * sizeof(short));
  float* pwout = ymean + TVEC_ELEMS;

  // conv1: 256*16*20*512 threads
  k_conv1<<<163840, 256, 0, stream>>>(x, conv1_w, conv1_b, bufA);

  static const int kindT_[12] = {0,0,1,0,0,1,0,0,0,0,0,0};
  static const int dil_[12]   = {1,1,2,2,2,4,4,4,4,8,8,8};
  static const int str_[12]   = {1,1,2,1,1,2,1,1,1,1,1,1};

  int Fr = 20;
  for (int i = 0; i < 12; ++i) {
    const float* const* bp = (const float* const*)(d_in + 3 + i * 11);
    const float* fdw = bp[0]; const float* tdw = bp[1]; const float* pw = bp[2];
    const float* sg = bp[3];  const float* sb = bp[4];
    const float* sm = bp[5];  const float* sv = bp[6];
    const float* bg = bp[7];  const float* bbi = bp[8];
    const float* bm = bp[9];  const float* bv = bp[10];
    int FrOut = Fr / str_[i];

    dim3 g2(4, 16, 256);
    k_freqdw<<<g2, 128, 0, stream>>>(bufA, bufB, ymean, fdw, sg, sb, sm, sv,
                                     Fr, FrOut, str_[i], kindT_[i]);
    // one workgroup per n, 8 waves, TDM-staged LDS, 4 WMMA tiles per wave
    k_temporal_pw<<<256, 256, 0, stream>>>(ymean, pwout, tdw, pw, bg, bbi, bm, bv,
                                           dil_[i]);
    long total = 256L * 16 * FrOut * 500;
    k_residual<<<(int)((total + 255) / 256), 256, 0, stream>>>(
        bufB, pwout, bufA, Fr, FrOut, kindT_[i]);
    Fr = FrOut;
  }
  k_final<<<256, 128, 0, stream>>>(bufA, conv4_w, (float*)d_out);
}
